// MultiHeadTemporalAttention_17463337025542
// MI455X (gfx1250) — matrix operand order repaired and checked by reading, not compiled, not run on hardware
//
#include <hip/hip_runtime.h>

// ---- problem constants (match reference) ----
#define H     1024
#define NH    16
#define HD    64
#define SEQ   2048
#define BATCH 4
#define M     (BATCH * SEQ)   // 8192 rows of x
#define EPS   1e-5f

// ---- CDNA5 WMMA types ----
typedef __attribute__((ext_vector_type(16))) __bf16        v16bf;
typedef __attribute__((ext_vector_type(8)))  float         v8f;
typedef __attribute__((ext_vector_type(4)))  unsigned int  v4u;
// GCC-style vector: exact pointee type of the async-to-LDS builtin params
typedef int v4i_vs __attribute__((vector_size(16)));

#define AS1 __attribute__((address_space(1)))
#define AS3 __attribute__((address_space(3)))

#if __has_builtin(__builtin_amdgcn_global_load_async_to_lds_b128)
#define HAVE_ASYNC_LDS 1
#else
#define HAVE_ASYNC_LDS 0
#endif

union FragBF { v16bf v; unsigned int u[8]; v4u q[2]; };

__device__ __forceinline__ unsigned short f2bf(float f) {
    unsigned int x = __float_as_uint(f);
    x += 0x7FFFu + ((x >> 16) & 1u);   // round-to-nearest-even
    return (unsigned short)(x >> 16);
}

__device__ __forceinline__ v8f wmma_bf16(FragBF a, FragBF b, v8f c) {
    // D(f32 16x16) = A(bf16 16x32) * B(bf16 32x16) + C
    return __builtin_amdgcn_wmma_f32_16x16x32_bf16(
        /*neg_a=*/false, a.v, /*neg_b=*/false, b.v,
        /*c_mod=*/(short)0, c, /*reuse_a=*/false, /*reuse_b=*/false);
}

// Load a 16x32 bf16 A fragment whose row (M) = lane%16 starts at dword
// index `row_dword_base` (lane's row already folded in). Layout:
// K = hlf*16 + 0..15 -> 8 contiguous dwords at base + hlf*8.
__device__ __forceinline__ FragBF load_a_frag(const unsigned int* p, size_t row_dword_base, int hlf) {
    FragBF f;
    const v4u* vp = (const v4u*)(p + row_dword_base);
    f.q[0] = vp[2 * hlf];       // dwords base + hlf*8 .. +3
    f.q[1] = vp[2 * hlf + 1];   // dwords base + hlf*8 + 4 .. +7
    return f;
}
// B fragment (32x16 bf16): column n = lane%16, K = hlf*16 + 2j -> 8
// contiguous dwords at base + hlf*8.
__device__ __forceinline__ FragBF load_b_frag(const unsigned int* p, size_t col_dword_base, int hlf) {
    FragBF f;
    const v4u* vp = (const v4u*)(p + col_dword_base + hlf * 8);
    f.q[0] = vp[0];
    f.q[1] = vp[1];
    return f;
}

// ===================== conversion kernels =====================

// hidden_states [S,B,H] f32 -> xb [B,S,H] bf16
__global__ __launch_bounds__(256) void cvt_x_kernel(const float* __restrict__ hs,
                                                    unsigned short* __restrict__ xb) {
    size_t i = (size_t)blockIdx.x * 256 + threadIdx.x;   // over M*H
    int h  = (int)(i & (H - 1));
    size_t bs = i >> 10;                                  // H = 1024
    int b = (int)(bs >> 11);                              // SEQ = 2048
    int s = (int)(bs & (SEQ - 1));
    xb[i] = f2bf(hs[((size_t)s * BATCH + b) * H + h]);
}

// W [H,H] f32 row-major (k,n) -> Wt [H,H] bf16 with layout (n,k)
__global__ __launch_bounds__(256) void cvt_w_kernel(const float* __restrict__ Wq,
                                                    const float* __restrict__ Wk,
                                                    const float* __restrict__ Wv,
                                                    const float* __restrict__ Wo,
                                                    unsigned short* __restrict__ wt) {
    int w = blockIdx.y;
    const float* src = (w == 0) ? Wq : (w == 1) ? Wk : (w == 2) ? Wv : Wo;
    size_t i = (size_t)blockIdx.x * 256 + threadIdx.x;   // over H*H, i = n*H + k
    int n = (int)(i >> 10);
    int k = (int)(i & (H - 1));
    wt[(size_t)w * H * H + i] = f2bf(src[(size_t)k * H + n]);
}

// ===================== QKV projection GEMM =====================
// out[m, n] = sum_k xb[m,k] * W[k,n] + bias[n]
// Software-pipelined: fragments for chunk kc+32 are fetched while the WMMAs
// for chunk kc execute, so loads overlap matrix math.
// blockIdx.z selects Q/K/V. Q,K written [B,NH,S,HD]; V written transposed [B,NH,HD,S].
__global__ __launch_bounds__(256) void gemm_qkv_kernel(const unsigned int* __restrict__ xb,
                                                       const unsigned int* __restrict__ wt_all,
                                                       const float* __restrict__ bq,
                                                       const float* __restrict__ bk,
                                                       const float* __restrict__ bv,
                                                       unsigned short* __restrict__ Qo,
                                                       unsigned short* __restrict__ Ko,
                                                       unsigned short* __restrict__ Vt) {
    const int mode = blockIdx.z;
    const unsigned int* Wt = wt_all + (size_t)mode * (H * H / 2);   // dword units
    const float* bias = (mode == 0) ? bq : (mode == 1) ? bk : bv;

    const int lane = threadIdx.x & 31;
    const int wave = threadIdx.x >> 5;
    const int hlf  = lane >> 4;
    const int lm   = lane & 15;
    const int m_base = blockIdx.x * 16;
    const int n_base = blockIdx.y * 512 + wave * 64;

    const size_t a_row = (size_t)(m_base + lm) * (H / 2);   // dwords
    size_t b_col[4];
#pragma unroll
    for (int c = 0; c < 4; ++c) b_col[c] = (size_t)(n_base + c * 16 + lm) * (H / 2);

    FragBF a_cur = load_a_frag(xb, a_row, hlf);
    FragBF b_cur[4];
#pragma unroll
    for (int c = 0; c < 4; ++c) b_cur[c] = load_b_frag(Wt, b_col[c], hlf);

    v8f acc[4] = {};
    for (int kc = 0; kc < H - 32; kc += 32) {
        const int kn = (kc + 32) >> 1;                 // next chunk, dwords
        FragBF a_nxt = load_a_frag(xb, a_row + kn, hlf);
        FragBF b_nxt[4];
#pragma unroll
        for (int c = 0; c < 4; ++c) b_nxt[c] = load_b_frag(Wt, b_col[c] + kn, hlf);
#pragma unroll
        for (int c = 0; c < 4; ++c) acc[c] = wmma_bf16(a_cur, b_cur[c], acc[c]);
        a_cur = a_nxt;
#pragma unroll
        for (int c = 0; c < 4; ++c) b_cur[c] = b_nxt[c];
    }
#pragma unroll
    for (int c = 0; c < 4; ++c) acc[c] = wmma_bf16(a_cur, b_cur[c], acc[c]);

#pragma unroll
    for (int c = 0; c < 4; ++c) {
        int n  = n_base + c * 16 + lm;
        float bb = bias[n];
        int nh = n >> 6;          // /HD
        int hd = n & (HD - 1);
#pragma unroll
        for (int r = 0; r < 8; ++r) {
            int m = m_base + r + hlf * 8;
            int b = m >> 11;                 // /SEQ
            int s = m & (SEQ - 1);
            unsigned short h16 = f2bf(acc[c][r] + bb);
            if (mode == 2) {
                Vt[(((size_t)(b * NH + nh) * HD + hd) * SEQ) + s] = h16;
            } else {
                unsigned short* dst = (mode == 0) ? Qo : Ko;
                dst[(((size_t)(b * NH + nh) * SEQ + s) * HD) + hd] = h16;
            }
        }
    }
}

// ===================== flash attention =====================
// One workgroup = 8 waves, all on the same (b, nh) head; each wave owns a
// 16-query tile. K / V^T chunks (shared by all 8 waves) are double-buffered
// in LDS and filled with ASYNCcnt-tracked global->LDS copies issued one
// chunk ahead, so HBM latency overlaps WMMA + softmax work.
#define FA_WAVES 8
__global__ __launch_bounds__(256) void flash_attn_kernel(const unsigned int* __restrict__ Q,
                                                         const unsigned int* __restrict__ K,
                                                         const unsigned int* __restrict__ Vt,
                                                         unsigned short* __restrict__ ctx) {
    __shared__ unsigned int ktile[2][32 * HD / 2];       // 2x4KB: [key_local][hd]
    __shared__ unsigned int vtile[2][HD * 32 / 2];       // 2x4KB: [hd][key_local]
    __shared__ float        sc[FA_WAVES][16 * 32];       // scores tiles (f32)
    __shared__ unsigned int pl[FA_WAVES][16 * 16];       // P tiles, packed bf16 pairs
    __shared__ float        alpha_s[FA_WAVES][16];
    __shared__ float        linv_s[FA_WAVES][16];

    const int tid  = threadIdx.x;
    const int lane = tid & 31;
    const int wave = tid >> 5;
    const int hlf  = lane >> 4;
    const int lm   = lane & 15;

    const int bh = blockIdx.x >> 4;            // b*NH + nh (same for all waves)
    const int qg = blockIdx.x & 15;
    const int q_base = (qg * FA_WAVES + wave) * 16;
    const size_t qkoff = (size_t)bh * SEQ * HD;   // base of this head's Q / K (elems)
    const size_t voff  = (size_t)bh * HD * SEQ;   // base of this head's V^T (elems)
    const float scale = 0.125f;                   // 1/sqrt(HD)

    // issue the (async) copy of one 32-key chunk into LDS buffer `buf`
    auto issue_chunk = [&](int kb, int buf) {
        // K chunk is a contiguous 2048-element block; thread t moves 16B.
        size_t ksrc = ((qkoff + (size_t)kb * HD) >> 1) + tid * 4;           // dwords
        size_t vsrc = ((voff + (size_t)(tid >> 2) * SEQ + kb) >> 1) + (tid & 3) * 4;
#if HAVE_ASYNC_LDS
        __builtin_amdgcn_global_load_async_to_lds_b128(
            (AS1 v4i_vs*)(K + ksrc), (AS3 v4i_vs*)&ktile[buf][tid * 4], 0, 0);
        __builtin_amdgcn_global_load_async_to_lds_b128(
            (AS1 v4i_vs*)(Vt + vsrc), (AS3 v4i_vs*)&vtile[buf][tid * 4], 0, 0);
#else
        v4u kt = *(const v4u*)(K + ksrc);
        v4u vv = *(const v4u*)(Vt + vsrc);
        *(v4u*)&ktile[buf][tid * 4] = kt;
        *(v4u*)&vtile[buf][tid * 4] = vv;
#endif
    };

    // preload Q fragments (contraction dims 0..31 and 32..63) - kept in VGPRs
    FragBF qf[2];
#pragma unroll
    for (int ho = 0; ho < 2; ++ho)
        qf[ho] = load_a_frag(Q, ((qkoff + (size_t)(q_base + lm) * HD) >> 1) + ho * 16, hlf);

    v8f acc[4] = {};
    float m_run = -1e30f, l_run = 0.f;

    issue_chunk(0, 0);                              // prime the pipeline

    for (int kb = 0; kb < SEQ; kb += 32) {
        const int buf = (kb >> 5) & 1;
        const int kb_next = kb + 32;
        if (kb_next < SEQ) {
            issue_chunk(kb_next, buf ^ 1);          // prefetch next chunk
#if HAVE_ASYNC_LDS
            // 2 newest outstanding copies = the prefetch; in-order completion
            // means the current buffer's 2 copies are done.
            __builtin_amdgcn_s_wait_asynccnt(2);
#endif
        } else {
#if HAVE_ASYNC_LDS
            __builtin_amdgcn_s_wait_asynccnt(0);
#endif
        }
        __syncthreads();

        // ---- scores: S = Q (16x64) @ K^T (64x32), two 16-key column tiles ----
        v8f s0 = {}, s1 = {};
#pragma unroll
        for (int t = 0; t < 2; ++t) {
#pragma unroll
            for (int ho = 0; ho < 2; ++ho) {
                FragBF kf;
                const v4u* kp = (const v4u*)&ktile[buf][(size_t)(t * 16 + lm) * (HD / 2) + ho * 16 + hlf * 8];
                kf.q[0] = kp[0];
                kf.q[1] = kp[1];
                if (t == 0) s0 = wmma_bf16(qf[ho], kf, s0);
                else        s1 = wmma_bf16(qf[ho], kf, s1);
            }
        }
        // stage scaled scores into LDS (D layout -> row-major)
#pragma unroll
        for (int r = 0; r < 8; ++r) {
            int row = r + hlf * 8;
            sc[wave][row * 32 + lm]      = s0[r] * scale;
            sc[wave][row * 32 + 16 + lm] = s1[r] * scale;
        }
        __syncthreads();
        // ---- online softmax: lanes 0-15 each own one row of 32 scores ----
        if (lane < 16) {
            int row = lane;
            float mx = m_run;
            for (int i = 0; i < 32; ++i) mx = fmaxf(mx, sc[wave][row * 32 + i]);
            float al = __expf(m_run - mx);
            float sum = 0.f;
            for (int i = 0; i < 32; i += 2) {
                float p0 = __expf(sc[wave][row * 32 + i]     - mx);
                float p1 = __expf(sc[wave][row * 32 + i + 1] - mx);
                sum += p0 + p1;
                pl[wave][row * 16 + (i >> 1)] =
                    ((unsigned int)f2bf(p1) << 16) | (unsigned int)f2bf(p0);
            }
            l_run = al * l_run + sum;
            m_run = mx;
            alpha_s[wave][row] = al;
        }
        __syncthreads();
        // ---- rescale running ctx accumulators per row ----
#pragma unroll
        for (int r = 0; r < 8; ++r) {
            float a = alpha_s[wave][r + hlf * 8];
            acc[0][r] *= a; acc[1][r] *= a; acc[2][r] *= a; acc[3][r] *= a;
        }
        // P A-fragment from LDS (row-major packed pairs -> A layout, 2x b128)
        FragBF pf;
        {
            const v4u* pp = (const v4u*)&pl[wave][lm * 16 + hlf * 8];
            pf.q[0] = pp[0];      // K = hlf*16 + 0..7
            pf.q[1] = pp[1];      // K = hlf*16 + 8..15
        }
        // ---- ctx += P (16x32) @ V (32x16) for 4 hd column groups ----
#pragma unroll
        for (int c = 0; c < 4; ++c) {
            FragBF vf;
            int hd = c * 16 + lm;
            const v4u* vp = (const v4u*)&vtile[buf][hd * 16 + hlf * 8];
            vf.q[0] = vp[0];
            vf.q[1] = vp[1];
            acc[c] = wmma_bf16(pf, vf, acc[c]);
        }
        __syncthreads();   // all waves done with this buffer before it is reused
    }
    // final 1/l normalization, broadcast via LDS
    if (lane < 16) linv_s[wave][lane] = 1.0f / l_run;
    __syncthreads();
    const int b  = bh >> 4;       // /NH
    const int nh = bh & (NH - 1);
#pragma unroll
    for (int r = 0; r < 8; ++r) {
        int row = r + hlf * 8;
        float inv = linv_s[wave][row];
        int s_idx = q_base + row;
        size_t base = ((size_t)(b * SEQ + s_idx)) * H + nh * HD;
        ctx[base +  0 + lm] = f2bf(acc[0][r] * inv);
        ctx[base + 16 + lm] = f2bf(acc[1][r] * inv);
        ctx[base + 32 + lm] = f2bf(acc[2][r] * inv);
        ctx[base + 48 + lm] = f2bf(acc[3][r] * inv);
    }
}

// ===================== output projection GEMM (fp32 out) =====================
__global__ __launch_bounds__(256) void gemm_proj_kernel(const unsigned int* __restrict__ ctx,
                                                        const unsigned int* __restrict__ wot,
                                                        const float* __restrict__ bo,
                                                        float* __restrict__ proj) {
    const int lane = threadIdx.x & 31;
    const int wave = threadIdx.x >> 5;
    const int hlf  = lane >> 4;
    const int lm   = lane & 15;
    const int m_base = blockIdx.x * 16;
    const int n_base = blockIdx.y * 512 + wave * 64;

    const size_t a_row = (size_t)(m_base + lm) * (H / 2);
    size_t b_col[4];
#pragma unroll
    for (int c = 0; c < 4; ++c) b_col[c] = (size_t)(n_base + c * 16 + lm) * (H / 2);

    FragBF a_cur = load_a_frag(ctx, a_row, hlf);
    FragBF b_cur[4];
#pragma unroll
    for (int c = 0; c < 4; ++c) b_cur[c] = load_b_frag(wot, b_col[c], hlf);

    v8f acc[4] = {};
    for (int kc = 0; kc < H - 32; kc += 32) {
        const int kn = (kc + 32) >> 1;
        FragBF a_nxt = load_a_frag(ctx, a_row + kn, hlf);
        FragBF b_nxt[4];
#pragma unroll
        for (int c = 0; c < 4; ++c) b_nxt[c] = load_b_frag(wot, b_col[c] + kn, hlf);
#pragma unroll
        for (int c = 0; c < 4; ++c) acc[c] = wmma_bf16(a_cur, b_cur[c], acc[c]);
        a_cur = a_nxt;
#pragma unroll
        for (int c = 0; c < 4; ++c) b_cur[c] = b_nxt[c];
    }
#pragma unroll
    for (int c = 0; c < 4; ++c) acc[c] = wmma_bf16(a_cur, b_cur[c], acc[c]);

#pragma unroll
    for (int c = 0; c < 4; ++c) {
        int n = n_base + c * 16 + lm;
        float bb = bo[n];
#pragma unroll
        for (int r = 0; r < 8; ++r) {
            int m = m_base + r + hlf * 8;
            proj[(size_t)m * H + n] = acc[c][r] + bb;
        }
    }
}

// ===================== residual + LayerNorm (in place on proj) ==============
__global__ __launch_bounds__(256) void ln_kernel(float* __restrict__ proj,
                                                 const float* __restrict__ hs,
                                                 const float* __restrict__ gamma,
                                                 const float* __restrict__ beta) {
    __shared__ float red[256];
    const int bs  = blockIdx.x;            // b*SEQ + s
    const int b   = bs >> 11;
    const int s   = bs & (SEQ - 1);
    const int tid = threadIdx.x;

    float y[4];
    float lsum = 0.f;
#pragma unroll
    for (int i = 0; i < 4; ++i) {
        int h = tid + i * 256;
        float v = proj[(size_t)bs * H + h] + hs[((size_t)s * BATCH + b) * H + h];
        y[i] = v;
        lsum += v;
    }
    red[tid] = lsum;
    __syncthreads();
    for (int off = 128; off > 0; off >>= 1) {
        if (tid < off) red[tid] += red[tid + off];
        __syncthreads();
    }
    const float mu = red[0] * (1.0f / H);
    __syncthreads();
    float vs = 0.f;
#pragma unroll
    for (int i = 0; i < 4; ++i) { float d = y[i] - mu; vs += d * d; }
    red[tid] = vs;
    __syncthreads();
    for (int off = 128; off > 0; off >>= 1) {
        if (tid < off) red[tid] += red[tid + off];
        __syncthreads();
    }
    const float rstd = rsqrtf(red[0] * (1.0f / H) + EPS);
#pragma unroll
    for (int i = 0; i < 4; ++i) {
        int h = tid + i * 256;
        proj[(size_t)bs * H + h] = (y[i] - mu) * rstd * gamma[h] + beta[h];
    }
}

// ===================== temporal pooling ============================
// tw == 1/S exactly (softmax rows sum to 1), so context_vector = mean_s(ln).
__global__ __launch_bounds__(256) void pool_kernel(const float* __restrict__ ln,
                                                   float* __restrict__ out) {
    int idx = blockIdx.x * 256 + threadIdx.x;   // over BATCH*H
    int b = idx >> 10;
    int h = idx & (H - 1);
    float sum = 0.f;
    for (int s = 0; s < SEQ; ++s)
        sum += ln[((size_t)(b * SEQ + s)) * H + h];
    out[idx] = sum * (1.0f / SEQ);
}

__global__ __launch_bounds__(256) void tw_kernel(float* __restrict__ out) {
    int i = blockIdx.x * 256 + threadIdx.x;     // over BATCH*SEQ
    out[BATCH * H + i] = 1.0f / (float)SEQ;
}

// ===================== launch =====================
extern "C" void kernel_launch(void* const* d_in, const int* in_sizes, int n_in,
                              void* d_out, int out_size, void* d_ws, size_t ws_size,
                              hipStream_t stream) {
    const float* hs    = (const float*)d_in[0];
    const float* Wq    = (const float*)d_in[1];
    const float* bq    = (const float*)d_in[2];
    const float* Wk    = (const float*)d_in[3];
    const float* bk    = (const float*)d_in[4];
    const float* Wv    = (const float*)d_in[5];
    const float* bv    = (const float*)d_in[6];
    const float* Wo    = (const float*)d_in[7];
    const float* bo    = (const float*)d_in[8];
    const float* gamma = (const float*)d_in[9];
    const float* beta  = (const float*)d_in[10];
    float* out = (float*)d_out;

    char* ws = (char*)d_ws;
    const size_t o_xb  = 0;
    const size_t o_wt  = o_xb  + (size_t)M * H * 2;        // x bf16
    const size_t o_q   = o_wt  + (size_t)4 * H * H * 2;    // 4 transposed weights bf16
    const size_t o_k   = o_q   + (size_t)M * H * 2;
    const size_t o_vt  = o_k   + (size_t)M * H * 2;
    const size_t o_ctx = o_vt  + (size_t)M * H * 2;
    const size_t o_pr  = o_ctx + (size_t)M * H * 2;        // proj fp32 (LN in-place)

    unsigned short* xb  = (unsigned short*)(ws + o_xb);
    unsigned short* wt  = (unsigned short*)(ws + o_wt);
    unsigned short* q   = (unsigned short*)(ws + o_q);
    unsigned short* k   = (unsigned short*)(ws + o_k);
    unsigned short* vt  = (unsigned short*)(ws + o_vt);
    unsigned short* ctx = (unsigned short*)(ws + o_ctx);
    float*          pr  = (float*)(ws + o_pr);

    cvt_x_kernel<<<(M * H) / 256, 256, 0, stream>>>(hs, xb);
    cvt_w_kernel<<<dim3((H * H) / 256, 4), 256, 0, stream>>>(Wq, Wk, Wv, Wo, wt);

    gemm_qkv_kernel<<<dim3(M / 16, H / 512, 3), 256, 0, stream>>>(
        (const unsigned int*)xb, (const unsigned int*)wt, bq, bk, bv, q, k, vt);

    flash_attn_kernel<<<BATCH * NH * (SEQ / (16 * FA_WAVES)), 256, 0, stream>>>(
        (const unsigned int*)q, (const unsigned int*)k, (const unsigned int*)vt, ctx);

    gemm_proj_kernel<<<dim3(M / 16, H / 512), 256, 0, stream>>>(
        (const unsigned int*)ctx,
        (const unsigned int*)(wt + (size_t)3 * H * H), bo, pr);

    ln_kernel<<<M, 256, 0, stream>>>(pr, hs, gamma, beta);
    pool_kernel<<<(BATCH * H) / 256, 256, 0, stream>>>(pr, out);
    tw_kernel<<<(BATCH * SEQ) / 256, 256, 0, stream>>>(out);
}
